// SNN1_27736898798056
// MI455X (gfx1250) — matrix operand order repaired and checked by reading, not compiled, not run on hardware
//
#include <hip/hip_runtime.h>

typedef _Float16 v4h  __attribute__((ext_vector_type(4)));
typedef _Float16 v8h  __attribute__((ext_vector_type(8)));
typedef _Float16 v16h __attribute__((ext_vector_type(16)));
typedef float    v4f  __attribute__((ext_vector_type(4)));
typedef float    v8f  __attribute__((ext_vector_type(8)));

#define B_  32
#define NI_ 1024
#define NH_ 2048
#define NO_ 512
#define T_  256

// ---------------------------------------------------------------------------
// Kernel 0: convert W1/W2 (f32) -> f16 in workspace
// ---------------------------------------------------------------------------
__global__ void wconv_kernel(const float* __restrict__ W1,
                             const float* __restrict__ W2,
                             _Float16* __restrict__ W1h,
                             _Float16* __restrict__ W2h) {
  long idx = (long)blockIdx.x * blockDim.x + threadIdx.x;
  long i4 = idx * 4;
  const long n1 = (long)NH_ * NI_;
  if (i4 < n1) {
    v4f v = *(const v4f*)(W1 + i4);
    v4h h = { (_Float16)v[0], (_Float16)v[1], (_Float16)v[2], (_Float16)v[3] };
    *(v4h*)(W1h + i4) = h;
  } else {
    long j = i4 - n1;
    v4f v = *(const v4f*)(W2 + j);
    v4h h = { (_Float16)v[0], (_Float16)v[1], (_Float16)v[2], (_Float16)v[3] };
    *(v4h*)(W2h + j) = h;
  }
}

// ---------------------------------------------------------------------------
// Kernel 1: LIF on input x [B][NI][T] (f32) -> spikes s1 [B][T][NI] (f16)
// Block: 256 threads = 256 neurons of one batch. LDS-staged x tiles
// (256 neurons x 16 t, stride 20 floats for bank spread), per-thread T scan.
// ---------------------------------------------------------------------------
__global__ void lif1_kernel(const float* __restrict__ x,
                            const float* __restrict__ beta1,
                            const float* __restrict__ thr1,
                            _Float16* __restrict__ s1) {
  __shared__ __align__(16) float xl[256 * 20];
  const int tid = threadIdx.x;
  const int b   = blockIdx.y;
  const int i0  = blockIdx.x * 256;
  const int i   = i0 + tid;

  // per-thread tile addresses (loop-invariant)
  const int r0 = tid >> 2, c0 = (tid & 3) * 4;
  const float* xsrc = x + ((long)(b * NI_ + i0 + r0)) * T_ + c0;

  float beta = beta1[i]; beta = fminf(fmaxf(beta, 0.0f), 1.0f);
  float thr  = thr1[i];
  float mem  = 0.0f;

  for (int tc = 0; tc < T_; tc += 16) {
    __syncthreads();
    // cooperative coalesced load of a [256 x 16] f32 tile: 4 loads in flight
    v4f n0 = *(const v4f*)(xsrc + tc);
    v4f n1 = *(const v4f*)(xsrc + (long)64 * T_ + tc);
    v4f n2 = *(const v4f*)(xsrc + (long)128 * T_ + tc);
    v4f n3 = *(const v4f*)(xsrc + (long)192 * T_ + tc);
    *(v4f*)(xl + (r0      ) * 20 + c0) = n0;
    *(v4f*)(xl + (r0 +  64) * 20 + c0) = n1;
    *(v4f*)(xl + (r0 + 128) * 20 + c0) = n2;
    *(v4f*)(xl + (r0 + 192) * 20 + c0) = n3;
    __syncthreads();
    #pragma unroll
    for (int c = 0; c < 16; ++c) {
      float xv    = xl[tid * 20 + c];
      float reset = (mem - thr) > 0.0f ? thr : 0.0f;
      mem = beta * mem + xv - reset;
      _Float16 spk = (mem - thr) > 0.0f ? (_Float16)1.0f : (_Float16)0.0f;
      s1[((long)b * T_ + tc + c) * NI_ + i] = spk;  // t-major, coalesced in i
    }
  }
}

// ---------------------------------------------------------------------------
// Kernels 2 & 3: fused GEMM (WMMA f16->f32) + LIF scan.
//   h[og][t] = sum_k W[og][k] * s_in[b][t][k]  + bias[og]   then LIF over t.
// Block = 256 threads (8 waves). Block tile = 64 rows x 256 t.
//   wave&3  -> 16-row M stripe, wave>>2 -> half of the time axis (8 n-tiles).
//   acc = 8 x v8f = 64 VGPRs/wave.
// B k-slab staged in LDS, DOUBLE-BUFFERED (2 x [256 t][40 halves] = 40 KB):
//   next slab's global loads go straight into 4 named uint4 registers before
//   the 8 WMMAs and are ds_stored to the alternate buffer after -> global
//   latency hides behind the WMMA chain, one barrier per k-step, no scratch.
// C fragments transposed via LDS hbuf[256][64] f32, then 64 threads scan T.
// SPIKES=true : write f16 spikes to outp [B][T][NOUTW]
// SPIKES=false: write f32 membrane to outp [B][NOUTW][T]  (d_out)
// ---------------------------------------------------------------------------
template<int K, int NOUTW, bool SPIKES>
__global__ __launch_bounds__(256, 1)
void gemm_lif_kernel(const _Float16* __restrict__ Wh,
                     const float* __restrict__ bias,
                     const float* __restrict__ betaA,
                     const float* __restrict__ thrA,
                     const _Float16* __restrict__ s_in,
                     void* __restrict__ outp) {
  __shared__ __align__(16) unsigned char smem[65536];
  constexpr int SLABB = 256 * 80;     // bytes per slab buffer ([256 t][80 B])
  float* hbuf = (float*)smem;         // [256 t][64 m] f32 (64 KB, post-loop)

  const int tid    = threadIdx.x;     // 0..255
  const int lane   = tid & 31;
  const int wave   = tid >> 5;        // 0..7
  const int mwave  = wave & 3;        // M stripe
  const int nthalf = wave >> 2;       // which half of the 16 n-tiles
  const int hi     = lane >> 4;       // 0/1
  const int l15    = lane & 15;
  const int b      = blockIdx.y;
  const int mbase  = blockIdx.x * 64;

  v8f acc[8];
  const v8f vzero = {0.f, 0.f, 0.f, 0.f, 0.f, 0.f, 0.f, 0.f};
  #pragma unroll
  for (int nt = 0; nt < 8; ++nt) acc[nt] = vzero;

  // A fragment row for this lane (ISA 16-bit A 16x32 layout)
  const _Float16* arow  = Wh + (long)(mbase + mwave * 16 + l15) * K;
  const _Float16* srow0 = s_in + (long)b * T_ * K;

  // per-thread slab-copy addresses (loop-invariant; only +kb / buffer varies)
  // thread covers rows t = tid>>2 (+64,+128,+192), 16B chunk c = tid&3
  const int  tr = tid >> 2;
  const int  cc = tid & 3;
  const _Float16* g0 = srow0 + (long)(tr      ) * K + cc * 8;
  const _Float16* g1 = srow0 + (long)(tr +  64) * K + cc * 8;
  const _Float16* g2 = srow0 + (long)(tr + 128) * K + cc * 8;
  const _Float16* g3 = srow0 + (long)(tr + 192) * K + cc * 8;
  const int d0 = (tr      ) * 80 + cc * 16;   // byte offsets in slab buffer
  const int d1 = (tr +  64) * 80 + cc * 16;
  const int d2 = (tr + 128) * 80 + cc * 16;
  const int d3 = (tr + 192) * 80 + cc * 16;

  // prologue: fill buffer 0 for kb = 0
  {
    uint4 n0 = *(const uint4*)(g0);
    uint4 n1 = *(const uint4*)(g1);
    uint4 n2 = *(const uint4*)(g2);
    uint4 n3 = *(const uint4*)(g3);
    char* dst = (char*)smem;
    *(uint4*)(dst + d0) = n0;
    *(uint4*)(dst + d1) = n1;
    *(uint4*)(dst + d2) = n2;
    *(uint4*)(dst + d3) = n3;
  }
  __syncthreads();

  int cur = 0;
  for (int kb = 0; kb < K; kb += 32) {
    const _Float16* sl = (const _Float16*)(smem + cur * SLABB);

    // A fragment first: halves 0..15 -> K = kb + 16*hi + 0..15
    const _Float16* ap = arow + kb + hi * 16;
    v8h alo = *(const v8h*)ap;
    v8h ahi = *(const v8h*)(ap + 8);
    v16h a = __builtin_shufflevector(alo, ahi,
             0,1,2,3,4,5,6,7,8,9,10,11,12,13,14,15);

    // issue next slab's global loads into named registers (no scratch)
    const bool more = (kb + 32) < K;
    uint4 n0, n1, n2, n3;
    if (more) {
      n0 = *(const uint4*)(g0 + kb + 32);
      n1 = *(const uint4*)(g1 + kb + 32);
      n2 = *(const uint4*)(g2 + kb + 32);
      n3 = *(const uint4*)(g3 + kb + 32);
    }

    // B fragment: col n = l15 (t = (nthalf*8+nt)*16 + l15),
    // halves K = kb + 16*hi + 0..15 (contiguous in t-major slab row)
    auto loadB = [&](int nt) -> v16h {
      const _Float16* bp = sl + ((nthalf * 8 + nt) * 16 + l15) * 40 + hi * 16;
      v8h blo = *(const v8h*)bp;
      v8h bhi = *(const v8h*)(bp + 8);
      return __builtin_shufflevector(blo, bhi,
             0,1,2,3,4,5,6,7,8,9,10,11,12,13,14,15);
    };

    v16h bcur = loadB(0);
    #pragma unroll
    for (int nt = 0; nt < 8; ++nt) {
      v16h bnxt = bcur;
      if (nt < 7) bnxt = loadB(nt + 1);
      acc[nt] = __builtin_amdgcn_wmma_f32_16x16x32_f16(
                  false, a, false, bcur, (short)0, acc[nt], false, false);
      bcur = bnxt;
    }

    // park next k-slab in the alternate buffer; single barrier per k-step
    if (more) {
      char* dst = (char*)smem + (cur ^ 1) * SLABB;
      *(uint4*)(dst + d0) = n0;
      *(uint4*)(dst + d1) = n1;
      *(uint4*)(dst + d2) = n2;
      *(uint4*)(dst + d3) = n3;
    }
    __syncthreads();
    cur ^= 1;
  }

  // transpose C fragments into hbuf[t][m]: vgpr r <-> m = mwave*16 + 8*hi + r
  const int mlo = mwave * 16 + hi * 8;
  #pragma unroll
  for (int nt = 0; nt < 8; ++nt) {
    int t = (nthalf * 8 + nt) * 16 + l15;
    float* hp = hbuf + t * 64 + mlo;
    *(v4f*)(hp)     = __builtin_shufflevector(acc[nt], acc[nt], 0, 1, 2, 3);
    *(v4f*)(hp + 4) = __builtin_shufflevector(acc[nt], acc[nt], 4, 5, 6, 7);
  }
  __syncthreads();

  // sequential LIF scan over t: 64 threads, one row each (conflict-free reads)
  if (tid < 64) {
    const int og = mbase + tid;
    float bsv  = bias[og];
    float beta = betaA[og]; beta = fminf(fmaxf(beta, 0.0f), 1.0f);
    float thr  = thrA[og];
    float mem  = 0.0f;
    if constexpr (SPIKES) {
      _Float16* sp = (_Float16*)outp + (long)b * T_ * NOUTW + og;
      for (int t = 0; t < T_; ++t) {
        float v     = hbuf[t * 64 + tid] + bsv;
        float reset = (mem - thr) > 0.0f ? thr : 0.0f;
        mem = beta * mem + v - reset;
        sp[(long)t * NOUTW] = (mem - thr) > 0.0f ? (_Float16)1.0f : (_Float16)0.0f;
      }
    } else {
      float* op = (float*)outp + ((long)b * NOUTW + og) * T_;
      v4f buf = {0.f, 0.f, 0.f, 0.f};
      for (int t = 0; t < T_; ++t) {
        float v     = hbuf[t * 64 + tid] + bsv;
        float reset = (mem - thr) > 0.0f ? thr : 0.0f;
        mem = beta * mem + v - reset;
        buf[t & 3] = mem;
        if ((t & 3) == 3) *(v4f*)(op + t - 3) = buf;
      }
    }
  }
}

// ---------------------------------------------------------------------------
extern "C" void kernel_launch(void* const* d_in, const int* in_sizes, int n_in,
                              void* d_out, int out_size, void* d_ws, size_t ws_size,
                              hipStream_t stream) {
  (void)in_sizes; (void)n_in; (void)out_size; (void)ws_size;
  const float* x     = (const float*)d_in[0];
  const float* beta1 = (const float*)d_in[1];
  const float* thr1  = (const float*)d_in[2];
  const float* W1    = (const float*)d_in[3];
  const float* b1    = (const float*)d_in[4];
  const float* beta2 = (const float*)d_in[5];
  const float* thr2  = (const float*)d_in[6];
  const float* W2    = (const float*)d_in[7];
  const float* b2    = (const float*)d_in[8];
  const float* beta3 = (const float*)d_in[9];
  const float* thr3  = (const float*)d_in[10];

  char* ws = (char*)d_ws;
  _Float16* W1h = (_Float16*)(ws + 0);                 //  4 MiB (NH*NI f16)
  _Float16* W2h = (_Float16*)(ws + 4194304);           //  2 MiB (NO*NH f16)
  _Float16* s1  = (_Float16*)(ws + 6291456);           // 16 MiB [B][T][NI] f16
  _Float16* s2  = (_Float16*)(ws + 23068672);          // 32 MiB [B][T][NH] f16
  float* out = (float*)d_out;                          // [B][NO][T] f32

  // W f32 -> f16
  wconv_kernel<<<3072, 256, 0, stream>>>(W1, W2, W1h, W2h);
  // LIF layer 1: x -> s1 (t-major spikes)
  lif1_kernel<<<dim3(NI_ / 256, B_), 256, 0, stream>>>(x, beta1, thr1, s1);
  // GEMM1 (M=2048,K=1024,N=256 xB) fused with LIF layer 2 -> s2
  gemm_lif_kernel<NI_, NH_, true>
      <<<dim3(NH_ / 64, B_), 256, 0, stream>>>(W1h, b1, beta2, thr2, s1, (void*)s2);
  // GEMM2 (M=512,K=2048,N=256 xB) fused with LIF layer 3 -> membrane out
  gemm_lif_kernel<NH_, NO_, false>
      <<<dim3(NO_ / 64, B_), 256, 0, stream>>>(W2h, b2, beta3, thr3, s2, (void*)out);
}
